// Comm_18279380812007
// MI455X (gfx1250) — matrix operand order repaired and checked by reading, not compiled, not run on hardware
//
#include <hip/hip_runtime.h>
#include <hip/hip_bf16.h>
#include <stdint.h>

#define H_   96
#define W_   192
#define HW_  (H_*W_)
#define C_   256
#define NC_  256
#define B_   4
#define N_   2
#define BIMG_ (B_*N_)

typedef __attribute__((ext_vector_type(16))) __bf16 v16bf;
typedef __attribute__((ext_vector_type(2)))  __bf16 v2bf;
typedef __attribute__((ext_vector_type(8)))  float  v8f;

union BFVec { uint4 q[2]; v16bf v; };
union U4S   { uint4 q; unsigned short s[8]; };

#if defined(__has_builtin)
#if __has_builtin(__builtin_amdgcn_cvt_pk_bf16_f32)
#define HAVE_PK_BF16 1
#endif
#endif

__device__ __forceinline__ unsigned short f2bf(float f) {
  unsigned u = __float_as_uint(f);
  u += 0x7FFFu + ((u >> 16) & 1u);           // round-to-nearest-even
  return (unsigned short)(u >> 16);
}
__device__ __forceinline__ unsigned pk2bf(float lo, float hi) {
#ifdef HAVE_PK_BF16
  union { v2bf v; unsigned u; } cvt;
  cvt.v = __builtin_amdgcn_cvt_pk_bf16_f32(lo, hi);
  return cvt.u;
#else
  return ((unsigned)f2bf(hi) << 16) | (unsigned)f2bf(lo);
#endif
}
__device__ __forceinline__ float bf2f(unsigned short h) {
  return __uint_as_float(((unsigned)h) << 16);
}
__device__ __forceinline__ float bflo(unsigned x) { return __uint_as_float(x << 16); }
__device__ __forceinline__ float bfhi(unsigned x) { return __uint_as_float(x & 0xFFFF0000u); }
__device__ __forceinline__ float fast_rcp(float x) { return __builtin_amdgcn_rcpf(x); }

// ---------------------------------------------------------------------------
// P0: bf16 copies of codebook (row-major + transposed) and norms
// ---------------------------------------------------------------------------
__global__ void __launch_bounds__(256)
prep_norms(const float* __restrict__ cb, const float* __restrict__ cbp,
           unsigned short* __restrict__ cbA, unsigned short* __restrict__ cbT,
           float* __restrict__ invn, float* __restrict__ rn, float* __restrict__ rnp) {
  __shared__ float sr[256];
  const int n = blockIdx.x, t = threadIdx.x;
  float v = cb[n*256 + t];
  cbA[n*256 + t] = f2bf(v);
  cbT[t*256 + n] = f2bf(v);
  sr[t] = v * v; __syncthreads();
  for (int s = 128; s > 0; s >>= 1) { if (t < s) sr[t] += sr[t+s]; __syncthreads(); }
  if (t == 0) { float n2 = sr[0]; invn[n] = 1.0f / n2; rn[n] = rsqrtf(n2); }
  __syncthreads();
  float pv = cbp[n*256 + t];
  sr[t] = pv * pv; __syncthreads();
  for (int s = 128; s > 0; s >>= 1) { if (t < s) sr[t] += sr[t+s]; __syncthreads(); }
  if (t == 0) rnp[n] = rsqrtf(sr[0]);
}

// ---------------------------------------------------------------------------
// P1: masked[n] = (max_j cos(cb[n], cbp[j]) <= 0.5)
// ---------------------------------------------------------------------------
__global__ void __launch_bounds__(256)
prep_mask(const float* __restrict__ cb, const float* __restrict__ cbp,
          const float* __restrict__ rn, const float* __restrict__ rnp,
          unsigned* __restrict__ msk) {
  __shared__ float scb[256];
  __shared__ float sred[256];
  const int n = blockIdx.x, t = threadIdx.x;
  scb[t] = cb[n*256 + t];
  __syncthreads();
  float dot = 0.f;
  const float* row = cbp + t*256;
  for (int c = 0; c < 256; ++c) dot += scb[c] * row[c];
  sred[t] = dot * rn[n] * rnp[t];
  __syncthreads();
  for (int s = 128; s > 0; s >>= 1) { if (t < s) sred[t] = fmaxf(sred[t], sred[t+s]); __syncthreads(); }
  if (t == 0) msk[n] = (sred[0] <= 0.5f) ? 1u : 0u;
}

// ---------------------------------------------------------------------------
// GEMM1: weights[bimg][pixel][code] = sigmoid((cb · feat)/||cb||^2), masked->0
// Workgroup tile: 64 codes x 32 pixels. K staged 64 channels per barrier;
// bf16 pairs packed into dword DS stores.
// ---------------------------------------------------------------------------
__global__ void __launch_bounds__(256)
gemm1_weights(const float* __restrict__ feat, const unsigned short* __restrict__ cbA,
              const float* __restrict__ invn, const unsigned* __restrict__ msk,
              unsigned short* __restrict__ Wpm) {
  __shared__ __align__(16) unsigned s_feat32[32*36];  // [pixel][64 k] bf16-pairs, padded
  __shared__ __align__(16) unsigned s_wout32[32*36];  // [pixel][64 code] bf16-pairs, padded
  __shared__ float    s_inv[64];
  __shared__ unsigned s_mk[64];

  const int t = threadIdx.x;
  const int lane = t & 31, wv = t >> 5;
  const int p0    = blockIdx.x * 32;
  const int cbase = blockIdx.y * 64;
  const int bimg  = blockIdx.z;
  const int agent = bimg & (N_ - 1);

  if (t < 64) { s_inv[t] = invn[cbase + t]; s_mk[t] = msk[cbase + t]; }

  const int ct = wv & 3;       // code sub-tile 0..3
  const int px = wv >> 2;      // pixel sub-tile 0..1
  const int lm = lane & 15, lh = lane >> 4;

  v8f acc = {};

  const int cp = t >> 3;          // channel pair 0..31 (two channels each)
  const int p4 = (t & 7) * 4;     // pixel quad

  const unsigned short* s_feat16 = reinterpret_cast<const unsigned short*>(s_feat32);
  const float* fbase = feat + (size_t)bimg*C_*HW_ + p0 + p4;

  #pragma unroll
  for (int kb = 0; kb < 4; ++kb) {
    const int k0 = kb * 64;
    const int ch = k0 + 2*cp;
    // stage + convert a [64 channels x 32 pixels] chunk, transposed to [pixel][k]
    float4 f0 = *reinterpret_cast<const float4*>(fbase + (size_t)(ch    )*HW_);
    float4 f1 = *reinterpret_cast<const float4*>(fbase + (size_t)(ch + 1)*HW_);
    if (kb < 3) __builtin_prefetch(fbase + (size_t)(ch + 64)*HW_, 0, 1);
    s_feat32[(p4+0)*36 + cp] = pk2bf(f0.x, f1.x);
    s_feat32[(p4+1)*36 + cp] = pk2bf(f0.y, f1.y);
    s_feat32[(p4+2)*36 + cp] = pk2bf(f0.z, f1.z);
    s_feat32[(p4+3)*36 + cp] = pk2bf(f0.w, f1.w);
    __syncthreads();

    #pragma unroll
    for (int kk = 0; kk < 2; ++kk) {
      BFVec a, b;
      // A-frag: codebook rows (M=codes), K split by lane-half
      const unsigned short* arow =
          cbA + (size_t)(cbase + ct*16 + lm)*256 + k0 + kk*32 + lh*8;
      a.q[0] = *reinterpret_cast<const uint4*>(arow);
      a.q[1] = *reinterpret_cast<const uint4*>(arow + 16);
      // B-frag: feature (K=channels x N=pixels), lane = pixel, K split by lane-half
      const unsigned short* brow = s_feat16 + (px*16 + lm)*72 + kk*32 + lh*8;
      b.q[0] = *reinterpret_cast<const uint4*>(brow);
      b.q[1] = *reinterpret_cast<const uint4*>(brow + 16);
      acc = __builtin_amdgcn_wmma_f32_16x16x32_bf16(false, a.v, false, b.v,
                                                    (short)0, acc, false, false);
    }
    __syncthreads();
  }

  // epilogue: sigmoid + mask, packed dword writes to LDS transpose buffer
  const int pixL  = px*16 + lm;           // C: N = lane%16
  const int cBase = ct*16 + 8*lh;         // C: M = r + 8*(lane>=16)
  #pragma unroll
  for (int rp = 0; rp < 4; ++rp) {
    float w0, w1;
    {
      const int codeL = cBase + 2*rp;
      float x = acc[2*rp] * s_inv[codeL];
      w0 = fast_rcp(1.0f + __expf(-x));
      if (agent > 0 && s_mk[codeL]) w0 = 0.0f;
    }
    {
      const int codeL = cBase + 2*rp + 1;
      float x = acc[2*rp+1] * s_inv[codeL];
      w1 = fast_rcp(1.0f + __expf(-x));
      if (agent > 0 && s_mk[codeL]) w1 = 0.0f;
    }
    s_wout32[pixL*36 + cBase/2 + rp] = pk2bf(w0, w1);
  }
  __syncthreads();

  // pixel-major store: Wpm[bimg][pixel][code]
  const int pix = t >> 3, chk = t & 7;
  uint4 v = *reinterpret_cast<const uint4*>(s_wout32 + pix*36 + chk*4);
  *reinterpret_cast<uint4*>(
      Wpm + ((size_t)(bimg*HW_ + p0 + pix))*NC_ + cbase + chk*8) = v;
}

// ---------------------------------------------------------------------------
// Final: warp-fill masked codes for agents>0, softmax over codes, GEMM2 back
// to channels via WMMA.  Workgroup = 32 pixels of one image.
// ---------------------------------------------------------------------------
__global__ void __launch_bounds__(256)
fuse_warp_softmax_combine(const unsigned short* __restrict__ Wpm,
                          const unsigned short* __restrict__ cbT,
                          const unsigned* __restrict__ msk,
                          const float* __restrict__ aff,
                          float* __restrict__ out) {
  __shared__ __align__(16) unsigned short s_w[32*264];   // [pixel][code] bf16, padded
  __shared__ __align__(16) float s_out[32*260];          // [pixel][chan] f32, padded
  __shared__ unsigned s_mk[256];
  __shared__ float s_pm[32][8];
  __shared__ float s_mx[32], s_si[32];

  const int t = threadIdx.x;
  const int p0   = blockIdx.x * 32;
  const int bimg = blockIdx.y;
  const int b = bimg >> 1, agent = bimg & 1;

  s_mk[t] = msk[t];

  const int pix = t >> 3, sub = t & 7;
  {
    const unsigned short* src = Wpm + ((size_t)(bimg*HW_ + p0 + pix))*NC_ + sub*32;
    uint4 v0 = *reinterpret_cast<const uint4*>(src);
    uint4 v1 = *reinterpret_cast<const uint4*>(src + 8);
    uint4 v2 = *reinterpret_cast<const uint4*>(src + 16);
    uint4 v3 = *reinterpret_cast<const uint4*>(src + 24);
    unsigned short* dst = s_w + pix*264 + sub*32;
    *reinterpret_cast<uint4*>(dst)      = v0;
    *reinterpret_cast<uint4*>(dst + 8)  = v1;
    *reinterpret_cast<uint4*>(dst + 16) = v2;
    *reinterpret_cast<uint4*>(dst + 24) = v3;
  }
  __syncthreads();

  if (agent > 0) {
    // theta = affine_matrix[b, agent, 0]  (layout [B][N][N][2][3])
    const int tb = (b*N_ + agent) * 12;
    const float t00 = aff[tb+0], t01 = aff[tb+1], t02 = aff[tb+2];
    const float t10 = aff[tb+3], t11 = aff[tb+4], t12 = aff[tb+5];
    const int pg = p0 + pix;
    const int hh = pg / W_, ww = pg % W_;
    const float gx = -1.f + 2.f * ww / (float)(W_ - 1);
    const float gy = -1.f + 2.f * hh / (float)(H_ - 1);
    const float xs = t00*gx + t01*gy + t02;
    const float ys = t10*gx + t11*gy + t12;
    const float x = (xs + 1.f) * 0.5f * (float)(W_ - 1);
    const float y = (ys + 1.f) * 0.5f * (float)(H_ - 1);
    const float x0f = floorf(x), y0f = floorf(y);
    const int ix0 = (int)x0f, iy0 = (int)y0f;
    const int ix1 = ix0 + 1, iy1 = iy0 + 1;
    const float fx = x - x0f, fy = y - y0f;
    const float wA = (1.f-fx)*(1.f-fy), wB = fx*(1.f-fy);
    const float wC = (1.f-fx)*fy,       wD = fx*fy;
    const bool vA = (ix0 >= 0) & (ix0 < W_) & (iy0 >= 0) & (iy0 < H_);
    const bool vB = (ix1 >= 0) & (ix1 < W_) & (iy0 >= 0) & (iy0 < H_);
    const bool vC = (ix0 >= 0) & (ix0 < W_) & (iy1 >= 0) & (iy1 < H_);
    const bool vD = (ix1 >= 0) & (ix1 < W_) & (iy1 >= 0) & (iy1 < H_);
    const unsigned short* ego = Wpm + ((size_t)(b*N_) * HW_) * NC_;  // agent-0 weights
    const uint4 z4 = make_uint4(0u, 0u, 0u, 0u);
    #pragma unroll
    for (int cc = 0; cc < 4; ++cc) {
      const int cb8 = sub*32 + cc*8;
      U4S a4, b4, c4, d4;
      a4.q = vA ? *reinterpret_cast<const uint4*>(ego + ((size_t)(iy0*W_+ix0))*NC_ + cb8) : z4;
      b4.q = vB ? *reinterpret_cast<const uint4*>(ego + ((size_t)(iy0*W_+ix1))*NC_ + cb8) : z4;
      c4.q = vC ? *reinterpret_cast<const uint4*>(ego + ((size_t)(iy1*W_+ix0))*NC_ + cb8) : z4;
      d4.q = vD ? *reinterpret_cast<const uint4*>(ego + ((size_t)(iy1*W_+ix1))*NC_ + cb8) : z4;
      #pragma unroll
      for (int j = 0; j < 8; ++j) {
        const int code = cb8 + j;
        if (s_mk[code]) {
          float v = wA*bf2f(a4.s[j]) + wB*bf2f(b4.s[j])
                  + wC*bf2f(c4.s[j]) + wD*bf2f(d4.s[j]);
          s_w[pix*264 + code] = f2bf(v);
        }
      }
    }
  }
  __syncthreads();

  // ---- softmax over 256 codes per pixel (8 threads/pixel) ----
  // cache this thread's 32-code slice (16 dwords) in registers
  unsigned qd[16];
  {
    const uint4* qp = reinterpret_cast<const uint4*>(s_w + pix*264 + sub*32);
    uint4 a0 = qp[0], a1 = qp[1], a2 = qp[2], a3 = qp[3];
    qd[0]=a0.x;  qd[1]=a0.y;  qd[2]=a0.z;  qd[3]=a0.w;
    qd[4]=a1.x;  qd[5]=a1.y;  qd[6]=a1.z;  qd[7]=a1.w;
    qd[8]=a2.x;  qd[9]=a2.y;  qd[10]=a2.z; qd[11]=a2.w;
    qd[12]=a3.x; qd[13]=a3.y; qd[14]=a3.z; qd[15]=a3.w;
  }
  float pm = -3.0e38f;
  #pragma unroll
  for (int i = 0; i < 16; ++i)
    pm = fmaxf(pm, fmaxf(bflo(qd[i]), bfhi(qd[i])));
  s_pm[pix][sub] = pm; __syncthreads();
  if (sub == 0) {
    float m0 = s_pm[pix][0];
    #pragma unroll
    for (int j = 1; j < 8; ++j) m0 = fmaxf(m0, s_pm[pix][j]);
    s_mx[pix] = m0;
  }
  __syncthreads();
  const float m = s_mx[pix];
  float ps = 0.f;
  #pragma unroll
  for (int i = 0; i < 16; ++i)
    ps += __expf(bflo(qd[i]) - m) + __expf(bfhi(qd[i]) - m);
  s_pm[pix][sub] = ps; __syncthreads();
  if (sub == 0) {
    float s = 0.f;
    #pragma unroll
    for (int j = 0; j < 8; ++j) s += s_pm[pix][j];
    s_si[pix] = fast_rcp(s);
  }
  __syncthreads();
  const float inv = s_si[pix];
  {
    unsigned* s_w32 = reinterpret_cast<unsigned*>(s_w);
    const int base32 = pix*132 + sub*16;
    #pragma unroll
    for (int i = 0; i < 16; ++i) {
      float e0 = __expf(bflo(qd[i]) - m) * inv;
      float e1 = __expf(bfhi(qd[i]) - m) * inv;
      s_w32[base32 + i] = pk2bf(e0, e1);
    }
  }
  __syncthreads();

  // ---- GEMM2: out[pixel][chan] = sum_code sw[pixel][code] * cbT[chan][code] ----
  const int lane = t & 31, wv = t >> 5;
  const int pxT = wv & 1;         // pixel sub-tile
  const int ntBase = wv >> 1;     // channel tile base 0..3
  const int lm = lane & 15, lh = lane >> 4;
  #pragma unroll
  for (int it = 0; it < 4; ++it) {
    const int nt = ntBase + it*4;    // channel tile 0..15
    v8f acc = {};
    #pragma unroll
    for (int kc = 0; kc < 8; ++kc) {
      const int k0 = kc * 32;
      BFVec a, bb;
      const unsigned short* arow = s_w + (pxT*16 + lm)*264 + k0 + lh*8;
      a.q[0] = *reinterpret_cast<const uint4*>(arow);
      a.q[1] = *reinterpret_cast<const uint4*>(arow + 16);
      const unsigned short* brow = cbT + (size_t)(nt*16 + lm)*256 + k0 + lh*8;
      bb.q[0] = *reinterpret_cast<const uint4*>(brow);
      bb.q[1] = *reinterpret_cast<const uint4*>(brow + 16);
      acc = __builtin_amdgcn_wmma_f32_16x16x32_bf16(false, a.v, false, bb.v,
                                                    (short)0, acc, false, false);
    }
    #pragma unroll
    for (int r = 0; r < 8; ++r) {
      const int pixL = pxT*16 + r + 8*lh;
      const int chL  = nt*16 + lm;
      s_out[pixL*260 + chL] = acc[r];
    }
  }
  __syncthreads();

  // ---- coalesced channel-major store: out[bimg][c][pixel] ----
  #pragma unroll
  for (int ccb = 0; ccb < 8; ++ccb) {
    const int c  = ccb*32 + (t >> 3);
    const int pj = (t & 7) * 4;
    float4 v;
    v.x = s_out[(pj+0)*260 + c];
    v.y = s_out[(pj+1)*260 + c];
    v.z = s_out[(pj+2)*260 + c];
    v.w = s_out[(pj+3)*260 + c];
    *reinterpret_cast<float4*>(out + ((size_t)(bimg*C_ + c))*HW_ + p0 + pj) = v;
  }
}

// ---------------------------------------------------------------------------
extern "C" void kernel_launch(void* const* d_in, const int* in_sizes, int n_in,
                              void* d_out, int out_size, void* d_ws, size_t ws_size,
                              hipStream_t stream) {
  const float* feat = (const float*)d_in[0];   // [8,256,96,192]
  const float* cb   = (const float*)d_in[1];   // [256,256]
  const float* cbp  = (const float*)d_in[2];   // [256,256]
  const float* aff  = (const float*)d_in[3];   // [4,2,2,2,3]
  (void)d_in[4]; (void)in_sizes; (void)n_in; (void)out_size; (void)ws_size;
  float* out = (float*)d_out;

  char* ws = (char*)d_ws;
  unsigned short* cbA  = (unsigned short*)(ws);               // 131072 B
  unsigned short* cbT  = (unsigned short*)(ws + 131072);      // 131072 B
  float*          invn = (float*)        (ws + 262144);       // 1024 B
  float*          rn   = (float*)        (ws + 263168);       // 1024 B
  float*          rnp  = (float*)        (ws + 264192);       // 1024 B
  unsigned*       msk  = (unsigned*)     (ws + 265216);       // 1024 B
  unsigned short* Wpm  = (unsigned short*)(ws + 266240);      // 75,497,472 B

  hipLaunchKernelGGL(prep_norms, dim3(256), dim3(256), 0, stream,
                     cb, cbp, cbA, cbT, invn, rn, rnp);
  hipLaunchKernelGGL(prep_mask, dim3(256), dim3(256), 0, stream,
                     cb, cbp, rn, rnp, msk);
  hipLaunchKernelGGL(gemm1_weights, dim3(HW_/32, 4, BIMG_), dim3(256), 0, stream,
                     feat, cbA, invn, msk, Wpm);
  hipLaunchKernelGGL(fuse_warp_softmax_combine, dim3(HW_/32, BIMG_), dim3(256), 0, stream,
                     Wpm, cbT, msk, aff, out);
}
